// NEE_18854906429830
// MI455X (gfx1250) — matrix operand order repaired and checked by reading, not compiled, not run on hardware
//
#include <hip/hip_runtime.h>
#include <math.h>
#include <stdint.h>

typedef __bf16 v16bf __attribute__((ext_vector_type(16)));
typedef __bf16 v8bf  __attribute__((ext_vector_type(8)));
typedef float  v8f   __attribute__((ext_vector_type(8)));

// ---------- constants (from reference) ----------
static constexpr int NN   = 50000;
static constexpr int EE   = 1000000;
static constexpr int DN   = 64;
static constexpr int E_IN = 3 * DN + 2 + DN;  // 258
static constexpr int E_KP = 288;              // padded to multiple of 32
static constexpr int HE   = 128;
static constexpr int OE   = 64;
static constexpr int N_IN = DN + OE;          // 128
static constexpr int HN   = 128;
static constexpr int ON   = 64;
static constexpr int ETILES = EE / 64;        // 15625
static constexpr int EDGE_BLOCKS = 4096;      // persistent blocks for edge kernel

// ---------- helpers ----------
__device__ __forceinline__ __bf16 f2bf(float x) {
  unsigned u = __builtin_bit_cast(unsigned, x);
  unsigned r = (u + 0x7FFFu + ((u >> 16) & 1u)) >> 16;
  unsigned short s = (unsigned short)r;
  return __builtin_bit_cast(__bf16, s);
}

__device__ __forceinline__ float gelu_exact(float y) {
  return 0.5f * y * (1.0f + erff(y * 0.70710678118654752f));
}

// CDNA5 async global->LDS copy (16B per active lane), tracked by ASYNCcnt.
// vdst = per-lane LDS byte offset (low 32 bits of flat LDS pointer), vaddr = 64-bit global addr.
__device__ __forceinline__ void async_g2l_b128(void* lds_dst, const void* gsrc) {
  unsigned lds_off = (unsigned)(uintptr_t)lds_dst;
  asm volatile("global_load_async_to_lds_b128 %0, %1, off"
               :: "v"(lds_off), "v"(gsrc)
               : "memory");
}
__device__ __forceinline__ void wait_async0() {
  asm volatile("s_wait_asynccnt 0x0" ::: "memory");
}

// C[64 x Ncols] = X[64 x Kdim] @ W  where Wt is W transposed: Wt[Ncols][Kdim] (bf16, K-contig).
// Block = 256 threads = 8 waves. Waves tiled 4(M) x 2(N-half); each wave: Ncols/32 n-tiles.
// A-frag (16-bit 16x32): lane(half=l/16, m=l%16): elems 0..7 -> K=kk+half*8+0..7,
//                        elems 8..15 -> K=kk+16+half*8+0..7   (two b128 LDS loads)
// B-frag (32x16): lane(half, n=l%16): elems 0..7 -> K=kk+half*8+0..7, elems 8..15 -> K=kk+16+half*8+0..7  (two b128 loads)
// C/D f32: VGPR v -> M=v+8*half, N=lane%16.
// K-loop unroll capped at 3 to stay under 256 VGPRs (2 workgroups / WGP residency).
// Internal __syncthreads() before store so C may alias X.
template <int Kdim, int Ncols>
__device__ __forceinline__ void gemm64(const __bf16* __restrict__ X,
                                       const __bf16* __restrict__ Wt,
                                       float* __restrict__ C) {
  constexpr int NT = Ncols / 32;
  const int t    = threadIdx.x;
  const int wave = t >> 5;
  const int lane = t & 31;
  const int half = lane >> 4;
  const int l16  = lane & 15;
  const int wm   = wave & 3;
  const int wn   = wave >> 2;
  const int m0   = wm * 16;

  v8f acc[NT];
#pragma unroll
  for (int i = 0; i < NT; ++i) acc[i] = (v8f)0.0f;

#pragma unroll 3
  for (int kk = 0; kk < Kdim; kk += 32) {
    union { v16bf v; v8bf h[2]; } ua;
    const __bf16* xp = X + (m0 + l16) * Kdim + kk + half * 8;
    ua.h[0] = *(const v8bf*)(xp);
    ua.h[1] = *(const v8bf*)(xp + 16);
#pragma unroll
    for (int nt = 0; nt < NT; ++nt) {
      const int n0 = wn * (Ncols / 2) + nt * 16;
      union { v16bf v; v8bf h[2]; } ub;
      const __bf16* wp = Wt + (n0 + l16) * Kdim + kk + half * 8;
      ub.h[0] = *(const v8bf*)(wp);
      ub.h[1] = *(const v8bf*)(wp + 16);
      acc[nt] = __builtin_amdgcn_wmma_f32_16x16x32_bf16(
          false, ua.v, false, ub.v, (short)0, acc[nt], false, false);
    }
  }
  __syncthreads();  // allow C to alias X
#pragma unroll
  for (int nt = 0; nt < NT; ++nt) {
    const int n0 = wn * (Ncols / 2) + nt * 16;
#pragma unroll
    for (int v = 0; v < 8; ++v)
      C[(m0 + half * 8 + v) * Ncols + (n0 + l16)] = acc[nt][v];
  }
}

// rows = 64, Ncols in {64,128}; 4 threads per row; (x+bias) -> LN(g,b) -> GELU -> bf16
template <int Ncols>
__device__ __forceinline__ void ln_gelu_bf16(const float* __restrict__ C,
                                             const float* __restrict__ bias,
                                             const float* __restrict__ gamma,
                                             const float* __restrict__ beta,
                                             __bf16* __restrict__ H) {
  constexpr int CPT = Ncols / 4;
  const int t = threadIdx.x;
  const int r = t >> 2, q = t & 3;
  const float* row = C + r * Ncols;
  float s = 0.f, ss = 0.f;
#pragma unroll
  for (int i = 0; i < CPT; ++i) {
    float x = row[q * CPT + i] + bias[q * CPT + i];
    s += x; ss += x * x;
  }
  s  += __shfl_xor(s, 1, 32);  ss += __shfl_xor(ss, 1, 32);
  s  += __shfl_xor(s, 2, 32);  ss += __shfl_xor(ss, 2, 32);
  const float mean = s * (1.0f / Ncols);
  const float var  = ss * (1.0f / Ncols) - mean * mean;
  const float rstd = rsqrtf(var + 1e-5f);
#pragma unroll
  for (int i = 0; i < CPT; ++i) {
    const int c = q * CPT + i;
    float x = row[c] + bias[c];
    float y = (x - mean) * rstd * gamma[c] + beta[c];
    H[r * Ncols + c] = f2bf(gelu_exact(y));
  }
}

// ---------- pre-kernels ----------
__global__ void zero_f32(float* __restrict__ p, int n) {
  int i = blockIdx.x * blockDim.x + threadIdx.x;
  if (i < n) p[i] = 0.0f;
}

// W[K][N] f32 row-major -> Wt[N][Kpad] bf16 (transposed, zero-padded K)
__global__ void transpose_pad_bf16(const float* __restrict__ W, __bf16* __restrict__ Wt,
                                   int K, int N, int Kpad) {
  int idx = blockIdx.x * blockDim.x + threadIdx.x;
  if (idx >= N * Kpad) return;
  int n = idx / Kpad, k = idx % Kpad;
  float v = (k < K) ? W[k * N + n] : 0.0f;
  Wt[idx] = f2bf(v);
}

// ---------- edge kernel: persistent blocks, 64 edges / tile ----------
__global__ __launch_bounds__(256, 4) void edge_kernel(
    const float* __restrict__ nf, const long long* __restrict__ eidx,
    const float* __restrict__ ef,
    const __bf16* __restrict__ W1t, const float* __restrict__ eb1,
    const float* __restrict__ eg1, const float* __restrict__ ebe1,
    const __bf16* __restrict__ W2t, const float* __restrict__ eb2,
    const float* __restrict__ eg2, const float* __restrict__ ebe2,
    float* __restrict__ e_out, float* __restrict__ agg) {
  extern __shared__ char smem[];
  __bf16* Xs  = (__bf16*)smem;                            // [64][288] bf16 = 36864 B
  float*  Cs  = (float*)smem;                             // alias: C1 [64][128] / C2 [64][64]
  __bf16* W1s = (__bf16*)(smem + 36864);                  // [128][288] = 73728 B
  __bf16* Hs  = (__bf16*)(smem + 36864 + 73728);          // [64][128]  = 16384 B
  __bf16* W2s = (__bf16*)(smem + 36864 + 73728 + 16384);  // [64][128]  = 16384 B
  __shared__ int   rowsS[64];
  __shared__ int   colsS[64];
  __shared__ float d2S[64];

  const int t = threadIdx.x;

  // stage weights into LDS once per (persistent) block via CDNA5 async global->LDS DMA
  {
    uint4* d1 = (uint4*)W1s; const uint4* s1 = (const uint4*)W1t;
    for (int i = t; i < (HE * E_KP * 2) / 16; i += 256) async_g2l_b128(d1 + i, s1 + i);
    uint4* d2 = (uint4*)W2s; const uint4* s2 = (const uint4*)W2t;
    for (int i = t; i < (OE * HE * 2) / 16; i += 256) async_g2l_b128(d2 + i, s2 + i);
  }

  for (long long tile = blockIdx.x; tile < ETILES; tile += gridDim.x) {
    const long long e0 = tile * 64;

    if (t < 64) {
      rowsS[t] = (int)eidx[e0 + t];
      colsS[t] = (int)eidx[(long long)EE + e0 + t];
      d2S[t]   = 0.0f;
    }
    __syncthreads();

    // build X = [src | tgt | diff | d | d2 | ef | pad] (bf16) + accumulate ||diff||^2
    for (int idx = t; idx < 64 * 64; idx += 256) {
      const int e = idx >> 6, k = idx & 63;
      const float s = nf[rowsS[e] * DN + k];
      const float g = nf[colsS[e] * DN + k];
      const float d = s - g;
      __bf16* xr = Xs + e * E_KP;
      xr[k]        = f2bf(s);
      xr[64 + k]   = f2bf(g);
      xr[128 + k]  = f2bf(d);
      xr[194 + k]  = f2bf(ef[(e0 + e) * DN + k]);
      __hip_atomic_fetch_add(&d2S[e], d * d, __ATOMIC_RELAXED, __HIP_MEMORY_SCOPE_WORKGROUP);
    }
    for (int idx = t; idx < 64 * 30; idx += 256) {        // pad cols 258..287
      const int e = idx / 30, k = idx % 30;
      Xs[e * E_KP + 258 + k] = f2bf(0.0f);
    }
    __syncthreads();
    if (t < 64) {
      const float d2 = d2S[t];
      Xs[t * E_KP + 192] = f2bf(sqrtf(d2));
      Xs[t * E_KP + 193] = f2bf(d2);
    }
    wait_async0();   // weights resident before first GEMM use (no-op after first tile)
    __syncthreads();

    gemm64<E_KP, HE>(Xs, W1s, Cs);          // internal barrier before aliased store
    __syncthreads();
    ln_gelu_bf16<HE>(Cs, eb1, eg1, ebe1, Hs);
    __syncthreads();
    gemm64<HE, OE>(Hs, W2s, Cs);
    __syncthreads();

    // final LN + GELU -> e_out, scatter-add to agg[col]
    {
      const int r = t >> 2, q = t & 3;
      const float* row = Cs + r * OE;
      float s = 0.f, ss = 0.f;
#pragma unroll
      for (int i = 0; i < 16; ++i) {
        float x = row[q * 16 + i] + eb2[q * 16 + i];
        s += x; ss += x * x;
      }
      s  += __shfl_xor(s, 1, 32);  ss += __shfl_xor(ss, 1, 32);
      s  += __shfl_xor(s, 2, 32);  ss += __shfl_xor(ss, 2, 32);
      const float mean = s * (1.0f / OE);
      const float var  = ss * (1.0f / OE) - mean * mean;
      const float rstd = rsqrtf(var + 1e-5f);
      const long long dst = colsS[r];
      float* eo = e_out + (e0 + r) * OE;
#pragma unroll
      for (int i = 0; i < 16; ++i) {
        const int c = q * 16 + i;
        float x = row[c] + eb2[c];
        float y = (x - mean) * rstd * eg2[c] + ebe2[c];
        float g = gelu_exact(y);
        eo[c] = g;
        __hip_atomic_fetch_add(&agg[dst * OE + c], g, __ATOMIC_RELAXED, __HIP_MEMORY_SCOPE_AGENT);
      }
    }
    __syncthreads();   // Cs/colsS consumed before next tile overwrites them
  }
}

// ---------- node kernel: 64 nodes / block ----------
__global__ __launch_bounds__(256, 4) void node_kernel(
    const float* __restrict__ nf, const float* __restrict__ agg,
    const __bf16* __restrict__ W1t, const float* __restrict__ nb1,
    const float* __restrict__ ng1, const float* __restrict__ nbe1,
    const __bf16* __restrict__ W2t, const float* __restrict__ nb2,
    const float* __restrict__ ng2, const float* __restrict__ nbe2,
    float* __restrict__ n_out, int nNodes) {
  extern __shared__ char smem[];
  __bf16* Xs  = (__bf16*)smem;                            // [64][128] bf16 = 16384
  __bf16* W1s = (__bf16*)(smem + 16384);                  // [128][128]     = 32768
  __bf16* Hs  = (__bf16*)(smem + 16384 + 32768);          // [64][128]      = 16384
  __bf16* W2s = (__bf16*)(smem + 16384 + 32768 + 16384);  // [64][128]      = 16384
  float*  Cs  = (float*)(smem + 16384 + 32768 + 16384 + 16384);  // [64][128] f32 = 32768

  const int t   = threadIdx.x;
  const int n0b = blockIdx.x * 64;

  // async-stage weights; overlaps with the X-build gathers below
  {
    uint4* d1 = (uint4*)W1s; const uint4* s1 = (const uint4*)W1t;
    for (int i = t; i < (HN * N_IN * 2) / 16; i += 256) async_g2l_b128(d1 + i, s1 + i);
    uint4* d2 = (uint4*)W2s; const uint4* s2 = (const uint4*)W2t;
    for (int i = t; i < (ON * HN * 2) / 16; i += 256) async_g2l_b128(d2 + i, s2 + i);
  }
  for (int idx = t; idx < 64 * N_IN; idx += 256) {
    const int r = idx >> 7, c = idx & 127;
    const int node = n0b + r;
    float v = 0.0f;
    if (node < nNodes) v = (c < DN) ? nf[node * DN + c] : agg[node * OE + (c - DN)];
    Xs[idx] = f2bf(v);
  }
  wait_async0();
  __syncthreads();

  gemm64<N_IN, HN>(Xs, W1s, Cs);
  __syncthreads();
  ln_gelu_bf16<HN>(Cs, nb1, ng1, nbe1, Hs);
  __syncthreads();
  gemm64<HN, ON>(Hs, W2s, Cs);
  __syncthreads();

  {
    const int r = t >> 2, q = t & 3;
    const float* row = Cs + r * ON;
    float s = 0.f, ss = 0.f;
#pragma unroll
    for (int i = 0; i < 16; ++i) {
      float x = row[q * 16 + i] + nb2[q * 16 + i];
      s += x; ss += x * x;
    }
    s  += __shfl_xor(s, 1, 32);  ss += __shfl_xor(ss, 1, 32);
    s  += __shfl_xor(s, 2, 32);  ss += __shfl_xor(ss, 2, 32);
    const float mean = s * (1.0f / ON);
    const float var  = ss * (1.0f / ON) - mean * mean;
    const float rstd = rsqrtf(var + 1e-5f);
    const int node = n0b + r;
    if (node < nNodes) {
      float* no = n_out + (long long)node * ON;
#pragma unroll
      for (int i = 0; i < 16; ++i) {
        const int c = q * 16 + i;
        float x = row[c] + nb2[c];
        float y = (x - mean) * rstd * ng2[c] + nbe2[c];
        no[c] = gelu_exact(y);
      }
    }
  }
}

// ---------- launch ----------
extern "C" void kernel_launch(void* const* d_in, const int* in_sizes, int n_in,
                              void* d_out, int out_size, void* d_ws, size_t ws_size,
                              hipStream_t stream) {
  const float*     nf   = (const float*)d_in[0];
  const long long* eidx = (const long long*)d_in[1];
  const float*     ef   = (const float*)d_in[2];
  const float* eW1  = (const float*)d_in[3];
  const float* eb1  = (const float*)d_in[4];
  const float* eg1  = (const float*)d_in[5];
  const float* ebe1 = (const float*)d_in[6];
  const float* eW2  = (const float*)d_in[7];
  const float* eb2  = (const float*)d_in[8];
  const float* eg2  = (const float*)d_in[9];
  const float* ebe2 = (const float*)d_in[10];
  const float* nW1  = (const float*)d_in[11];
  const float* nb1  = (const float*)d_in[12];
  const float* ng1  = (const float*)d_in[13];
  const float* nbe1 = (const float*)d_in[14];
  const float* nW2  = (const float*)d_in[15];
  const float* nb2  = (const float*)d_in[16];
  const float* ng2  = (const float*)d_in[17];
  const float* nbe2 = (const float*)d_in[18];

  // workspace layout
  char* ws = (char*)d_ws;
  float*  agg   = (float*)ws;                                  // NN*OE*4 = 12,800,000 B
  size_t  off   = (size_t)NN * OE * sizeof(float);
  __bf16* eW1t  = (__bf16*)(ws + off);  off += (size_t)HE * E_KP * 2;   // 73728
  __bf16* eW2t  = (__bf16*)(ws + off);  off += (size_t)OE * HE * 2;     // 16384
  __bf16* nW1t  = (__bf16*)(ws + off);  off += (size_t)HN * N_IN * 2;   // 32768
  __bf16* nW2t  = (__bf16*)(ws + off);  off += (size_t)ON * HN * 2;     // 16384

  float* n_out = (float*)d_out;
  float* e_out = (float*)d_out + (size_t)NN * ON;

  zero_f32<<<(NN * OE + 255) / 256, 256, 0, stream>>>(agg, NN * OE);
  transpose_pad_bf16<<<(HE * E_KP + 255) / 256, 256, 0, stream>>>(eW1, eW1t, E_IN, HE, E_KP);
  transpose_pad_bf16<<<(OE * HE + 255) / 256, 256, 0, stream>>>(eW2, eW2t, HE, OE, HE);
  transpose_pad_bf16<<<(HN * N_IN + 255) / 256, 256, 0, stream>>>(nW1, nW1t, N_IN, HN, N_IN);
  transpose_pad_bf16<<<(ON * HN + 255) / 256, 256, 0, stream>>>(nW2, nW2t, HN, ON, HN);

  const size_t edge_lds = 36864 + 73728 + 16384 + 16384;   // 143,360 B
  edge_kernel<<<EDGE_BLOCKS, 256, edge_lds, stream>>>(
      nf, eidx, ef, eW1t, eb1, eg1, ebe1, eW2t, eb2, eg2, ebe2, e_out, agg);

  const size_t node_lds = 16384 + 32768 + 16384 + 16384 + 32768;  // 114,688 B
  node_kernel<<<(NN + 63) / 64, 256, node_lds, stream>>>(
      nf, agg, nW1t, nb1, ng1, nbe1, nW2t, nb2, ng2, nbe2, n_out, NN);
}
